// SCA_Block_16484084483400
// MI455X (gfx1250) — matrix operand order repaired and checked by reading, not compiled, not run on hardware
//
#include <hip/hip_runtime.h>
#include <hip/hip_bf16.h>
#include <cstdint>
#include <cstddef>

// ---------------------------------------------------------------------------
// SCA block for MI455X (gfx1250, wave32, WMMA).
//
// With P = WA*X, Q = WB*X, R = WE*X (bias-free):
//   scores[d,e] = (P Q^T)[d,e] + bA[d]*sumQ[e] + bB[e]*sumP[d] + HW*bA[d]*bB[e]
//   Emean[e]    = sumR[e]/HW + bE[e]
//   pooled[d]   = sum_e softmax_d(scores)[d,e] * Emean[e]
//   gate[c]     = sigmoid(pooled . WL[c,:] + bL[c]);  out = X * gate
// Heavy work = three [32x512]x[512x3136] bf16 GEMMs per batch + 32x32 Gram,
// all via v_wmma_f32_16x16x32_bf16; row sums fall out of the accumulators.
// ---------------------------------------------------------------------------

typedef __attribute__((ext_vector_type(16))) __bf16          v16bf;
typedef __attribute__((ext_vector_type(16))) unsigned short  v16us;
typedef __attribute__((ext_vector_type(8)))  float           v8f;

#define IN_C   512
#define DS_C   32
#define HWSZ   3136
#define NCHUNK 98          // 3136 / 32
#define BATCH  32
#define LDSS   40          // 32 + 8 pad (bf16 elems): row stride 80B = 5*16B

__device__ __forceinline__ unsigned short f2bf(float f) {
  unsigned u = __builtin_bit_cast(unsigned, f);
  u += 0x7FFFu + ((u >> 16) & 1u);
  return (unsigned short)(u >> 16);
}

// pack two f32 -> one dword of two bf16.  HW packed cvt if the builtin exists;
// otherwise 2x round-bias add + one v_perm_b32 (3 VALU per pair).
__device__ __forceinline__ unsigned pk2bf(float a, float b) {
#if __has_builtin(__builtin_amdgcn_cvt_pk_bf16_f32)
  auto r = __builtin_amdgcn_cvt_pk_bf16_f32(a, b);
  return __builtin_bit_cast(unsigned, r);
#else
  unsigned ua = __builtin_bit_cast(unsigned, a) + 0x7FFFu;   // nearest (ties-down)
  unsigned ub = __builtin_bit_cast(unsigned, b) + 0x7FFFu;
  // result bytes {ub[3],ub[2],ua[3],ua[2]} = (hi16(ub)<<16) | hi16(ua)
  return __builtin_amdgcn_perm(ub, ua, 0x07060302u);
#endif
}

__device__ __forceinline__ v8f wmma_bf16(v16us a, v16us b, v8f c) {
  return __builtin_amdgcn_wmma_f32_16x16x32_bf16(
      false, __builtin_bit_cast(v16bf, a),
      false, __builtin_bit_cast(v16bf, b),
      (short)0, c, false, false);
}

union VU { v16us v; unsigned q[8]; uint4 u4[2]; };

// ---------------------------------------------------------------------------
// Kernel 1: pack WA/WB/WE (f32 [32][512]) into bf16 A-operand fragment order.
// A-operand 16x32 bf16, contiguous k order used here (same for B): lane L holds row M=L&15,
// K(j) = ((L>=16)?16:0) + j.  Fragment (kt,m) = 32 lanes x 16
// bf16 contiguous -> one coalesced 32B load per lane in the GEMM kernel.
// ---------------------------------------------------------------------------
__global__ void sca_pack(const float* __restrict__ WA, const float* __restrict__ WB,
                         const float* __restrict__ WE,
                         unsigned short* __restrict__ pA, unsigned short* __restrict__ pB,
                         unsigned short* __restrict__ pE) {
  int o    = blockIdx.x * 256 + threadIdx.x;     // 0 .. 3*16384-1, exact
  int j    = o & 15;
  int lane = (o >> 4) & 31;
  int m    = (o >> 9) & 1;
  int kt   = (o >> 10) & 15;
  int mat  = o >> 14;
  const float* W    = (mat == 0) ? WA : (mat == 1) ? WB : WE;
  unsigned short* P = (mat == 0) ? pA : (mat == 1) ? pB : pE;
  int d = m * 16 + (lane & 15);
  int c = kt * 32 + ((lane >= 16) ? 16 : 0) + j;
  P[o & 16383] = f2bf(W[d * IN_C + c]);
}

// ---------------------------------------------------------------------------
// Kernel 2: per-batch projections + Gram via WMMA.
// Grid (8, 32): blockIdx.y = batch, 4 waves/WG, wave owns chunks of 32 hw.
// ---------------------------------------------------------------------------
__global__ __launch_bounds__(128) void sca_gemm(
    const float* __restrict__ X,
    const unsigned short* __restrict__ WpA, const unsigned short* __restrict__ WpB,
    const unsigned short* __restrict__ WpE,
    float* __restrict__ scores_raw,   // [32][32][32]  (P Q^T, no bias)
    float* __restrict__ sumP, float* __restrict__ sumQ, float* __restrict__ sumR) {
  __shared__ unsigned short lds[4 * 2 * 32 * LDSS];   // per-wave P-tile + Q-tile

  const int b    = blockIdx.y;
  const int g    = blockIdx.x;
  const int lane = threadIdx.x & 31;
  const int wave = threadIdx.x >> 5;
  const int lh   = lane & 15;
  const bool hi  = lane >= 16;

  unsigned short* At = lds + wave * (2 * 32 * LDSS);  // P tile [32 d][LDSS hw]
  unsigned short* Bt = At + 32 * LDSS;                // Q tile [32 e][LDSS hw]

  const float* Xb = X + (size_t)b * IN_C * HWSZ;

  v8f sc[2][2];                     // scores accumulators (d-tile, e-tile)
  v8f rsP[2], rsQ[2], rsR[2];       // running row-sums of P, Q, R
#pragma unroll
  for (int i = 0; i < 2; ++i) {
    sc[i][0] = (v8f)0.f; sc[i][1] = (v8f)0.f;
    rsP[i] = (v8f)0.f; rsQ[i] = (v8f)0.f; rsR[i] = (v8f)0.f;
  }

  for (int ck = g * 4 + wave; ck < NCHUNK; ck += 32) {   // wave-uniform: EXEC full
    const int hw0 = ck * 32;
    v8f aP[2][2], aQ[2][2], aR[2][2];
#pragma unroll
    for (int i = 0; i < 2; ++i)
#pragma unroll
      for (int n = 0; n < 2; ++n) { aP[i][n] = (v8f)0.f; aQ[i][n] = (v8f)0.f; aR[i][n] = (v8f)0.f; }

#pragma unroll 2
    for (int kt = 0; kt < 16; ++kt) {
      // --- weight fragments: one coalesced 32B load each (2x b128) ---
      const int fo = (kt * 2 * 32 + lane) * 16;
      v16us wa0 = *(const v16us*)(WpA + fo);
      v16us wa1 = *(const v16us*)(WpA + fo + 512);
      v16us wb0 = *(const v16us*)(WpB + fo);
      v16us wb1 = *(const v16us*)(WpB + fo + 512);
      v16us we0 = *(const v16us*)(WpE + fo);
      v16us we1 = *(const v16us*)(WpE + fo + 512);

      // --- X fragments, B-operand 32x16 layout: lane N=lh, K(j)=(hi?16:0)+j ---
      const float* xc = Xb + (size_t)(kt * 32 + (hi ? 16 : 0)) * HWSZ + hw0 + lh;
      __builtin_prefetch(xc + 1024, 0, 1);   // this wave's next chunk
      float f0[16], f1[16];
#pragma unroll
      for (int j = 0; j < 16; ++j) {
        f0[j] = xc[(size_t)j * HWSZ];
        f1[j] = xc[(size_t)j * HWSZ + 16];
      }
      VU ux0, ux1;
#pragma unroll
      for (int v = 0; v < 8; ++v) {          // 3 VALU per bf16 pair (v_perm_b32)
        ux0.q[v] = pk2bf(f0[2 * v], f0[2 * v + 1]);
        ux1.q[v] = pk2bf(f1[2 * v], f1[2 * v + 1]);
      }
      const v16us x0 = ux0.v, x1 = ux1.v;

      aP[0][0] = wmma_bf16(wa0, x0, aP[0][0]);
      aP[0][1] = wmma_bf16(wa0, x1, aP[0][1]);
      aP[1][0] = wmma_bf16(wa1, x0, aP[1][0]);
      aP[1][1] = wmma_bf16(wa1, x1, aP[1][1]);
      aQ[0][0] = wmma_bf16(wb0, x0, aQ[0][0]);
      aQ[0][1] = wmma_bf16(wb0, x1, aQ[0][1]);
      aQ[1][0] = wmma_bf16(wb1, x0, aQ[1][0]);
      aQ[1][1] = wmma_bf16(wb1, x1, aQ[1][1]);
      aR[0][0] = wmma_bf16(we0, x0, aR[0][0]);
      aR[0][1] = wmma_bf16(we0, x1, aR[0][1]);
      aR[1][0] = wmma_bf16(we1, x0, aR[1][0]);
      aR[1][1] = wmma_bf16(we1, x1, aR[1][1]);
    }

    // running row-sums over hw (final cross-lane reduce at end)
#pragma unroll
    for (int m = 0; m < 2; ++m) {
      rsP[m] += aP[m][0] + aP[m][1];
      rsQ[m] += aQ[m][0] + aQ[m][1];
      rsR[m] += aR[m][0] + aR[m][1];
    }

    // --- stage P, Q as bf16 in per-wave LDS to re-layout for the Gram ---
    // C/D layout: VGPR r, lane n<16 -> (M=r, N=n); n>=16 -> (M=r+8, N=n-16)
#pragma unroll
    for (int mt = 0; mt < 2; ++mt)
#pragma unroll
      for (int nt = 0; nt < 2; ++nt) {
        const int row0 = 16 * mt + (hi ? 8 : 0);
        const int h    = 16 * nt + lh;
#pragma unroll
        for (int r = 0; r < 8; r += 2) {
          unsigned pP = pk2bf(aP[mt][nt][r], aP[mt][nt][r + 1]);
          unsigned pQ = pk2bf(aQ[mt][nt][r], aQ[mt][nt][r + 1]);
          At[(row0 + r) * LDSS + h]     = (unsigned short)pP;
          At[(row0 + r + 1) * LDSS + h] = (unsigned short)(pP >> 16);
          Bt[(row0 + r) * LDSS + h]     = (unsigned short)pQ;
          Bt[(row0 + r + 1) * LDSS + h] = (unsigned short)(pQ >> 16);
        }
      }
    asm volatile("s_wait_dscnt 0" ::: "memory");   // stores visible before reads

    // GEMM2: scores += P(32d x 32hw) * Q^T(32hw x 32e), K = 32 (hw).
    // Fragments are byte-contiguous in the padded tiles -> 2x ds_load_b128.
    v16us af[2], bfm[2];
#pragma unroll
    for (int mt = 0; mt < 2; ++mt) {
      const unsigned short* rp = At + (16 * mt + lh) * LDSS + (hi ? 16 : 0);
      VU u;                                   // A-operand K = contiguous 32B
      u.u4[0] = *(const uint4*)(rp);
      u.u4[1] = *(const uint4*)(rp + 8);
      af[mt] = u.v;
    }
#pragma unroll
    for (int nt = 0; nt < 2; ++nt) {
      const unsigned short* rp = Bt + (16 * nt + lh) * LDSS + (hi ? 16 : 0);
      VU u;                                   // B-operand K = contiguous 32B
      u.u4[0] = *(const uint4*)(rp);
      u.u4[1] = *(const uint4*)(rp + 8);
      bfm[nt] = u.v;
    }
    sc[0][0] = wmma_bf16(af[0], bfm[0], sc[0][0]);
    sc[0][1] = wmma_bf16(af[0], bfm[1], sc[0][1]);
    sc[1][0] = wmma_bf16(af[1], bfm[0], sc[1][0]);
    sc[1][1] = wmma_bf16(af[1], bfm[1], sc[1][1]);
    // no second wait: per-wave DS ops retire in order (DScnt), WAR is safe
  }

  // --- flush partial scores ---
#pragma unroll
  for (int mt = 0; mt < 2; ++mt)
#pragma unroll
    for (int nt = 0; nt < 2; ++nt) {
      const int e  = 16 * nt + lh;
      const int d0 = 16 * mt + (hi ? 8 : 0);
#pragma unroll
      for (int r = 0; r < 8; ++r)
        atomicAdd(&scores_raw[((size_t)b * 32 + d0 + r) * 32 + e], sc[mt][nt][r]);
    }

  // --- flush row sums: reduce across each 16-lane half, then atomic ---
  auto flush = [&](v8f r0, v8f r1, float* __restrict__ gs) {
#pragma unroll
    for (int m = 0; m < 2; ++m) {
      v8f rv = (m == 0) ? r0 : r1;
#pragma unroll
      for (int r = 0; r < 8; ++r) {
        float v = rv[r];
        v += __shfl_xor(v, 8, 32);
        v += __shfl_xor(v, 4, 32);
        v += __shfl_xor(v, 2, 32);
        v += __shfl_xor(v, 1, 32);
        if ((lane & 15) == 0)
          atomicAdd(&gs[b * 32 + 16 * m + r + (hi ? 8 : 0)], v);
      }
    }
  };
  flush(rsP[0], rsP[1], sumP);
  flush(rsQ[0], rsQ[1], sumQ);
  flush(rsR[0], rsR[1], sumR);
}

// ---------------------------------------------------------------------------
// Kernel 3: biases + softmax(axis=d) + pooled + sigmoid gate.  One block/batch.
// ---------------------------------------------------------------------------
__global__ __launch_bounds__(64) void sca_gate(
    const float* __restrict__ scores_raw,
    const float* __restrict__ sumP, const float* __restrict__ sumQ,
    const float* __restrict__ sumR,
    const float* __restrict__ bA, const float* __restrict__ bB,
    const float* __restrict__ bE,
    const float* __restrict__ WL, const float* __restrict__ bL,
    float* __restrict__ gate) {
  const int b = blockIdx.x;
  const int t = threadIdx.x;
  __shared__ float Z[32][32];
  __shared__ float Em[32];
  __shared__ float pooled[32];

  if (t < 32) {
    const int e = t;
    const float sQe = sumQ[b * 32 + e];
    const float bBe = bB[e];
    float col[32];
    float mx = -INFINITY;
#pragma unroll
    for (int d = 0; d < 32; ++d) {
      float v = scores_raw[((size_t)b * 32 + d) * 32 + e]
              + bA[d] * sQe + bBe * sumP[b * 32 + d]
              + (float)HWSZ * bA[d] * bBe;
      col[d] = v;
      mx = fmaxf(mx, v);
    }
    float s = 0.f;
#pragma unroll
    for (int d = 0; d < 32; ++d) { col[d] = __expf(col[d] - mx); s += col[d]; }
    const float inv = 1.f / s;
#pragma unroll
    for (int d = 0; d < 32; ++d) Z[d][e] = col[d] * inv;
    Em[e] = sumR[b * 32 + e] * (1.f / (float)HWSZ) + bE[e];
  }
  __syncthreads();
  if (t < 32) {
    float p = 0.f;
#pragma unroll
    for (int e = 0; e < 32; ++e) p += Z[t][e] * Em[e];
    pooled[t] = p;
  }
  __syncthreads();
  for (int c = t; c < IN_C; c += 64) {
    float a = bL[c];
#pragma unroll
    for (int d = 0; d < 32; ++d) a += pooled[d] * WL[c * 32 + d];
    gate[b * IN_C + c] = 1.f / (1.f + __expf(-a));
  }
}

// ---------------------------------------------------------------------------
// Kernel 4: out = feature_in * gate[b,c]   (float4 streaming, 410 MB pass)
// ---------------------------------------------------------------------------
__global__ __launch_bounds__(256) void sca_scale(const float4* __restrict__ X,
                                                 const float* __restrict__ gate,
                                                 float4* __restrict__ out, int n4) {
  int i = blockIdx.x * 256 + threadIdx.x;
  if (i >= n4) return;
  const float g = gate[i / (HWSZ / 4)];   // i/784 == b*512 + c
  float4 v = X[i];
  v.x *= g; v.y *= g; v.z *= g; v.w *= g;
  out[i] = v;
}

// ---------------------------------------------------------------------------
extern "C" void kernel_launch(void* const* d_in, const int* in_sizes, int n_in,
                              void* d_out, int out_size, void* d_ws, size_t ws_size,
                              hipStream_t stream) {
  const float* X  = (const float*)d_in[0];
  const float* WA = (const float*)d_in[1];
  const float* bA = (const float*)d_in[2];
  const float* WB = (const float*)d_in[3];
  const float* bB = (const float*)d_in[4];
  const float* WE = (const float*)d_in[5];
  const float* bE = (const float*)d_in[6];
  const float* WL = (const float*)d_in[7];
  const float* bL = (const float*)d_in[8];

  char* ws = (char*)d_ws;
  unsigned short* pA = (unsigned short*)(ws);              // 32 KB
  unsigned short* pB = (unsigned short*)(ws + 32768);      // 32 KB
  unsigned short* pE = (unsigned short*)(ws + 65536);      // 32 KB
  float* scores = (float*)(ws + 98304);                    // 128 KB
  float* sP     = (float*)(ws + 229376);                   // 4 KB
  float* sQ     = (float*)(ws + 233472);                   // 4 KB
  float* sR     = (float*)(ws + 237568);                   // 4 KB
  float* gate   = (float*)(ws + 241664);                   // 64 KB

  hipMemsetAsync(ws + 98304, 0, 131072 + 3 * 4096, stream);  // zero accumulators

  sca_pack<<<192, 256, 0, stream>>>(WA, WB, WE, pA, pB, pE);
  sca_gemm<<<dim3(8, BATCH), 128, 0, stream>>>(X, pA, pB, pE, scores, sP, sQ, sR);
  sca_gate<<<BATCH, 64, 0, stream>>>(scores, sP, sQ, sR, bA, bB, bE, WL, bL, gate);

  const int n4 = BATCH * IN_C * HWSZ / 4;
  sca_scale<<<(n4 + 255) / 256, 256, 0, stream>>>((const float4*)X, gate,
                                                  (float4*)d_out, n4);
}
